// CausalSelfAttention_49615462204147
// MI455X (gfx1250) — matrix operand order repaired and checked by reading, not compiled, not run on hardware
//
#include <hip/hip_runtime.h>

// ---------------------------------------------------------------------------
// Causal MHA for MI455X (gfx1250, wave32). All matmuls use
// v_wmma_f32_16x16x32_f16. Pipeline:
//   1) convert_x: x->f16 (vectorized);  transpose: weights->f16 Wt[n][k]
//      via coalesced 32x32 LDS tile transpose
//   2) gemm: Q = x*Wq (scaled 1/8), K = x*Wk, V = x*Wv (V stored transposed)
//   3) flash attention (causal, online softmax, LPT-ordered q-blocks)
//   4) gemm: out = y*Wo (fp32)
// ---------------------------------------------------------------------------

typedef __attribute__((ext_vector_type(16))) _Float16 v16h;
typedef __attribute__((ext_vector_type(8)))  float    v8f;

#define DIM   1024
#define NHEAD 16
#define HDIM  64
#define BATCH 2
#define SEQ   2048
#define ROWS  (BATCH * SEQ)       // 4096

union FragU { v16h h; uint4 u[2]; };

// A fragment, 16x32 f16 (MxK), source row-major with leading dim `ld` (halves).
// lane: m = lane%16, hl = lane/16; halves j<8 -> k = 8*hl + j,
// j>=8 -> k = 16 + 8*hl + (j-8). Two 16B loads.
__device__ __forceinline__ v16h load_afrag(const _Float16* base, int ld) {
  const int lane = threadIdx.x & 31;
  const int m = lane & 15, hl = lane >> 4;
  const _Float16* p = base + m * ld + 8 * hl;
  FragU f;
  f.u[0] = *(const uint4*)(p);
  f.u[1] = *(const uint4*)(p + 16);
  return f.h;
}

// B fragment, 32x16 f16 (KxN), source is the TRANSPOSED matrix Bt[n][k] with
// leading dim `ld`. lane: n = lane%16, hl = lane/16; halves j<8 -> k = 8*hl + j,
// j>=8 -> k = 16 + 8*hl + (j-8). Two 16B loads.
__device__ __forceinline__ v16h load_bfrag(const _Float16* baseT, int ld) {
  const int lane = threadIdx.x & 31;
  const int n = lane & 15, hl = lane >> 4;
  const _Float16* p = baseT + n * ld + 8 * hl;
  FragU f;
  f.u[0] = *(const uint4*)(p);
  f.u[1] = *(const uint4*)(p + 16);
  return f.h;
}

__device__ __forceinline__ v8f wmma_f16(v16h a, v16h b, v8f c) {
  return __builtin_amdgcn_wmma_f32_16x16x32_f16(false, a, false, b,
                                                (short)0, c, false, false);
}

// ---------------------------------------------------------------------------
// x -> f16, vectorized: float4 in, 4 packed halves (8B) out.
// Launch exactly 4194304/4 threads = 4096 blocks x 256.
// ---------------------------------------------------------------------------
__global__ __launch_bounds__(256)
void convert_x_kernel(const float* __restrict__ x, _Float16* __restrict__ xh) {
  const int i = (blockIdx.x * blockDim.x + threadIdx.x) * 4;
  const float4 v = *(const float4*)(x + i);
  union { _Float16 h[4]; uint2 u; } o;
  o.h[0] = (_Float16)v.x;
  o.h[1] = (_Float16)v.y;
  o.h[2] = (_Float16)v.z;
  o.h[3] = (_Float16)v.w;
  *(uint2*)(xh + i) = o.u;
}

// ---------------------------------------------------------------------------
// Coalesced transpose + downconvert: Wt[n*1024+k] = (f16)W[k*1024+n].
// Wq/Wk/Wv flatten to [k=d][n=h*64+hd]; Wo to [k=h*64+hd][n=d]; all 1024x1024.
// 32x32 tiles via LDS (padded stride 34 halves -> 17 banks, conflict-free).
// Grid: (32, 32, 4), block 256 (32x8).
// ---------------------------------------------------------------------------
__global__ __launch_bounds__(256)
void transpose_kernel(const float* __restrict__ Wq, const float* __restrict__ Wk,
                      const float* __restrict__ Wv, const float* __restrict__ Wo,
                      _Float16* __restrict__ wqt, _Float16* __restrict__ wkt,
                      _Float16* __restrict__ wvt, _Float16* __restrict__ wot) {
  __shared__ _Float16 tile[32][34];
  const float* W;
  _Float16* Wt;
  switch (blockIdx.z) {
    case 0:  W = Wq; Wt = wqt; break;
    case 1:  W = Wk; Wt = wkt; break;
    case 2:  W = Wv; Wt = wvt; break;
    default: W = Wo; Wt = wot; break;
  }
  const int tx = threadIdx.x & 31;      // fast dim
  const int ty = threadIdx.x >> 5;      // 0..7
  const int n0 = blockIdx.x * 32;
  const int k0 = blockIdx.y * 32;
#pragma unroll
  for (int r = 0; r < 4; ++r) {
    const int kr = ty + 8 * r;
    tile[kr][tx] = (_Float16)W[(k0 + kr) * DIM + n0 + tx];  // coalesced read
  }
  __syncthreads();
#pragma unroll
  for (int r = 0; r < 4; ++r) {
    const int nr = ty + 8 * r;
    Wt[(n0 + nr) * DIM + k0 + tx] = tile[tx][nr];           // coalesced write
  }
}

// ---------------------------------------------------------------------------
// Generic f16 GEMM: C[4096x1024] = A[4096x1024] * Bt^T, one wave per 32x64
// output tile (2x4 WMMA tiles), K stepped by 32. Store modes:
//   0: f16, layout [b,h,l,hd] (Q with scale, K)
//   2: f16, transposed layout [b,h,hd,l] (V)
//   3: f32 row-major (final projection)
// Grid must be 256 blocks x 256 threads (2048 waves = 128 row x 16 col tiles).
// ---------------------------------------------------------------------------
__global__ __launch_bounds__(256)
void gemm_kernel(const _Float16* __restrict__ A,
                 const _Float16* __restrict__ Bt,
                 void* __restrict__ out, int mode, float scale) {
  const int wid = (blockIdx.x * blockDim.x + threadIdx.x) >> 5;
  const int rt = wid >> 4;          // 0..127
  const int ct = wid & 15;          // 0..15
  const int r0 = rt * 32;
  const int c0 = ct * 64;

  v8f acc[2][4] = {};

  for (int kk = 0; kk < DIM; kk += 32) {
    v16h a0 = load_afrag(A + (r0 +  0) * DIM + kk, DIM);
    v16h a1 = load_afrag(A + (r0 + 16) * DIM + kk, DIM);
#pragma unroll
    for (int nt = 0; nt < 4; ++nt) {
      v16h b = load_bfrag(Bt + (c0 + nt * 16) * DIM + kk, DIM);
      acc[0][nt] = wmma_f16(a0, b, acc[0][nt]);
      acc[1][nt] = wmma_f16(a1, b, acc[1][nt]);
    }
  }

  const int lane = threadIdx.x & 31;
  const int nlo = lane & 15, hl = lane >> 4;

#pragma unroll
  for (int mt = 0; mt < 2; ++mt) {
#pragma unroll
    for (int nt = 0; nt < 4; ++nt) {
#pragma unroll
      for (int v = 0; v < 8; ++v) {
        const int m = r0 + mt * 16 + v + 8 * hl;   // global row (b*SEQ + l)
        const int n = c0 + nt * 16 + nlo;          // global col
        const float val = acc[mt][nt][v] * scale;
        if (mode == 3) {
          ((float*)out)[m * DIM + n] = val;
        } else {
          const int b = m >> 11, l = m & (SEQ - 1);
          const int h = n >> 6,  hd = n & (HDIM - 1);
          if (mode == 2) {  // V transposed: [b,h,hd,l]
            ((_Float16*)out)[((b * NHEAD + h) * HDIM + hd) * SEQ + l] =
                (_Float16)val;
          } else {          // Q/K: [b,h,l,hd]
            ((_Float16*)out)[((b * NHEAD + h) * SEQ + l) * HDIM + hd] =
                (_Float16)val;
          }
        }
      }
    }
  }
}

// ---------------------------------------------------------------------------
// Flash attention (causal). One wave per (b,h,q-block of 32 rows).
// Q pre-scaled by 1/sqrt(hd). K is [b,h,l,hd] (row-major => already the
// transposed-B layout for Q*K^T). V is [b,h,hd,l] (transposed-B for P*V).
// P transits through a per-wave LDS slab to convert C-layout -> A-layout.
// q-blocks are issued in REVERSE order (LPT): heaviest causal blocks first,
// light ones backfill the dispatch tail.
// Grid must be 256 blocks x 256 threads (2048 waves = 32 bh x 64 q-blocks).
// ---------------------------------------------------------------------------
__global__ __launch_bounds__(256)
void attn_kernel(const _Float16* __restrict__ Q,
                 const _Float16* __restrict__ K,
                 const _Float16* __restrict__ Vt,
                 _Float16* __restrict__ Y) {
  __shared__ __align__(16) _Float16 lds[8][32 * 32];

  const int wave = threadIdx.x >> 5;
  const int task = blockIdx.x * 8 + wave;
  const int bh = task >> 6;               // 0..31
  const int q0 = (63 - (task & 63)) * 32; // LPT: heavy q-blocks first
  const int lane = threadIdx.x & 31;
  const int nlo = lane & 15, hl = lane >> 4;

  const _Float16* Qb = Q  + (size_t)bh * SEQ * HDIM;
  const _Float16* Kb = K  + (size_t)bh * SEQ * HDIM;
  const _Float16* Vb = Vt + (size_t)bh * HDIM * SEQ;
  _Float16* myP = &lds[wave][0];

  // Preload Q fragments: 2 m-tiles x 2 k-steps (hd 0..31, 32..63)
  v16h qf[2][2];
#pragma unroll
  for (int mt = 0; mt < 2; ++mt)
#pragma unroll
    for (int ks = 0; ks < 2; ++ks)
      qf[mt][ks] = load_afrag(Qb + (q0 + mt * 16) * HDIM + ks * 32, HDIM);

  v8f acc[2][4] = {};
  float mrun[2][8], lsum[2][8];
#pragma unroll
  for (int mt = 0; mt < 2; ++mt)
#pragma unroll
    for (int v = 0; v < 8; ++v) { mrun[mt][v] = -3.0e38f; lsum[mt][v] = 0.f; }

  for (int k0 = 0; k0 <= q0 + 31; k0 += 32) {
    // ---- S = Q * K^T  (32q x 32k) ----
    v8f s[2][2] = {};
#pragma unroll
    for (int ks = 0; ks < 2; ++ks) {
      v16h b0 = load_bfrag(Kb + (k0 +  0) * HDIM + ks * 32, HDIM);
      v16h b1 = load_bfrag(Kb + (k0 + 16) * HDIM + ks * 32, HDIM);
      s[0][0] = wmma_f16(qf[0][ks], b0, s[0][0]);
      s[0][1] = wmma_f16(qf[0][ks], b1, s[0][1]);
      s[1][0] = wmma_f16(qf[1][ks], b0, s[1][0]);
      s[1][1] = wmma_f16(qf[1][ks], b1, s[1][1]);
    }

    // ---- causal mask (only the block touching the diagonal) ----
    if (k0 + 31 > q0) {
#pragma unroll
      for (int mt = 0; mt < 2; ++mt)
#pragma unroll
        for (int nt = 0; nt < 2; ++nt)
#pragma unroll
          for (int v = 0; v < 8; ++v) {
            const int qrow = q0 + mt * 16 + v + 8 * hl;
            const int key  = k0 + nt * 16 + nlo;
            if (key > qrow) s[mt][nt][v] = -3.0e38f;
          }
    }

    // ---- online softmax (row stats live replicated in each 16-lane half) ----
#pragma unroll
    for (int mt = 0; mt < 2; ++mt) {
#pragma unroll
      for (int v = 0; v < 8; ++v) {
        float t = fmaxf(s[mt][0][v], s[mt][1][v]);
        t = fmaxf(t, __shfl_xor(t, 1));
        t = fmaxf(t, __shfl_xor(t, 2));
        t = fmaxf(t, __shfl_xor(t, 4));
        t = fmaxf(t, __shfl_xor(t, 8));
        const float mnew = fmaxf(mrun[mt][v], t);
        const float corr = __expf(mrun[mt][v] - mnew);
        mrun[mt][v] = mnew;
        lsum[mt][v] *= corr;
#pragma unroll
        for (int nt = 0; nt < 4; ++nt) acc[mt][nt][v] *= corr;
        const float p0 = __expf(s[mt][0][v] - mnew);
        const float p1 = __expf(s[mt][1][v] - mnew);
        s[mt][0][v] = p0;
        s[mt][1][v] = p1;
        float ps = p0 + p1;
        ps += __shfl_xor(ps, 1);
        ps += __shfl_xor(ps, 2);
        ps += __shfl_xor(ps, 4);
        ps += __shfl_xor(ps, 8);
        lsum[mt][v] += ps;
      }
    }

    // ---- P: C-layout -> row-major LDS slab -> A-fragments ----
#pragma unroll
    for (int mt = 0; mt < 2; ++mt)
#pragma unroll
      for (int nt = 0; nt < 2; ++nt)
#pragma unroll
        for (int v = 0; v < 8; ++v) {
          const int ml = mt * 16 + v + 8 * hl;
          const int kl = nt * 16 + nlo;
          myP[ml * 32 + kl] = (_Float16)s[mt][nt][v];
        }
    asm volatile("s_wait_dscnt 0" ::: "memory");

    v16h pf0 = load_afrag(myP +  0 * 32, 32);
    v16h pf1 = load_afrag(myP + 16 * 32, 32);
    asm volatile("s_wait_dscnt 0" ::: "memory");

    // ---- O += P * V ----
#pragma unroll
    for (int nt = 0; nt < 4; ++nt) {
      v16h vf = load_bfrag(Vb + (nt * 16) * SEQ + k0, SEQ);
      acc[0][nt] = wmma_f16(pf0, vf, acc[0][nt]);
      acc[1][nt] = wmma_f16(pf1, vf, acc[1][nt]);
    }
  }

  // ---- normalize and write y[b,l,h*64+hd] (f16, row-major 4096x1024) ----
  const int h = bh & (NHEAD - 1), b = bh >> 4;
#pragma unroll
  for (int mt = 0; mt < 2; ++mt)
#pragma unroll
    for (int nt = 0; nt < 4; ++nt)
#pragma unroll
      for (int v = 0; v < 8; ++v) {
        const float val = acc[mt][nt][v] / lsum[mt][v];
        const int l = q0 + mt * 16 + v + 8 * hl;
        const int col = h * HDIM + nt * 16 + nlo;
        Y[(size_t)(b * SEQ + l) * DIM + col] = (_Float16)val;
      }
}

// ---------------------------------------------------------------------------
// Host launcher. Workspace layout (bytes), total 40 MiB:
//   xh (aliased by yh after V projection): 0        .. 8388608
//   wqt: 8388608   wkt: 10485760   wvt: 12582912   wot: 14680064
//   qh:  16777216  kh:  25165824   vt:  33554432   end: 41943040
// ---------------------------------------------------------------------------
extern "C" void kernel_launch(void* const* d_in, const int* in_sizes, int n_in,
                              void* d_out, int out_size, void* d_ws,
                              size_t ws_size, hipStream_t stream) {
  const float* x  = (const float*)d_in[0];
  const float* Wq = (const float*)d_in[1];
  const float* Wk = (const float*)d_in[2];
  const float* Wv = (const float*)d_in[3];
  const float* Wo = (const float*)d_in[4];
  float* out = (float*)d_out;

  char* ws = (char*)d_ws;
  _Float16* xh  = (_Float16*)(ws + 0);
  _Float16* wqt = (_Float16*)(ws + 8388608);
  _Float16* wkt = (_Float16*)(ws + 10485760);
  _Float16* wvt = (_Float16*)(ws + 12582912);
  _Float16* wot = (_Float16*)(ws + 14680064);
  _Float16* qh  = (_Float16*)(ws + 16777216);
  _Float16* kh  = (_Float16*)(ws + 25165824);
  _Float16* vt  = (_Float16*)(ws + 33554432);
  _Float16* yh  = xh;  // xh is dead after the V projection

  convert_x_kernel<<<4096, 256, 0, stream>>>(x, xh);
  transpose_kernel<<<dim3(32, 32, 4), 256, 0, stream>>>(Wq, Wk, Wv, Wo,
                                                        wqt, wkt, wvt, wot);
  // Q with 1/sqrt(HEAD_DIM) folded in
  gemm_kernel<<<256, 256, 0, stream>>>(xh, wqt, (void*)qh, 0, 0.125f);
  gemm_kernel<<<256, 256, 0, stream>>>(xh, wkt, (void*)kh, 0, 1.0f);
  gemm_kernel<<<256, 256, 0, stream>>>(xh, wvt, (void*)vt, 2, 1.0f);
  attn_kernel<<<256, 256, 0, stream>>>(qh, kh, vt, yh);
  gemm_kernel<<<256, 256, 0, stream>>>(yh, wot, (void*)out, 3, 1.0f);
}
